// GINModel_3521873183180
// MI455X (gfx1250) — matrix operand order repaired and checked by reading, not compiled, not run on hardware
//
#include <hip/hip_runtime.h>

#define NN 50000
#define NE 800000
#define DD 256
#define NL 5
#define NC 10
#define BN_EPS 1e-5f

typedef __attribute__((ext_vector_type(16))) __bf16 v16bf;
typedef __attribute__((ext_vector_type(8)))  float  v8f;

// ---------------- utility ----------------
__global__ void zero_f32(float* __restrict__ p, long n) {
  long i = (long)blockIdx.x * blockDim.x + threadIdx.x;
  long stride = (long)gridDim.x * blockDim.x;
  for (; i < n; i += stride) p[i] = 0.0f;
}

// ---------------- edge scatter-add (GIN aggregation) ----------------
// one wave per edge; lane l handles features l, l+32, ..., l+224
__global__ __launch_bounds__(256)
void scatter_add_kernel(const float* __restrict__ h,
                        const long long* __restrict__ src,
                        const long long* __restrict__ dst,
                        float* __restrict__ agg) {
  int wave = (blockIdx.x * blockDim.x + threadIdx.x) >> 5;
  int lane = threadIdx.x & 31;
  if (wave >= NE) return;
  int s = (int)src[wave];
  int d = (int)dst[wave];
  const float* hrow = h + s * DD;
  float* arow = agg + d * DD;
#pragma unroll
  for (int j = 0; j < DD / 32; ++j) {
    int f = lane + 32 * j;
    unsafeAtomicAdd(&arow[f], hrow[f]);
  }
}

// ---------------- fused WMMA GEMM ----------------
// MODE 0: A = h + agg
// MODE 1: A = (y1 - mean)*istd*gamma + beta   (BN applied on the fly)
// Computes Y = A @ W + bias, and accumulates per-column sum / sumsq of Y.
template <int MODE>
__global__ __launch_bounds__(256)
void gin_gemm_kernel(const float* __restrict__ A0,
                     const float* __restrict__ A1,
                     const float* __restrict__ mean,
                     const float* __restrict__ istd,
                     const float* __restrict__ gamma,
                     const float* __restrict__ beta,
                     const float* __restrict__ W,     // [DD][DD] row-major (k, n)
                     const float* __restrict__ bias,  // [DD]
                     float* __restrict__ Y,           // [NN][DD]
                     float* __restrict__ colsum,
                     float* __restrict__ colsumsq) {
  __shared__ __bf16 As[128][32];   // [m][k]
  __shared__ __bf16 Bs[64][32];    // [n][k]  (transposed for frag loads)

  const int mBase = blockIdx.x * 128;
  const int nBase = blockIdx.y * 64;
  const int tid   = threadIdx.x;
  const int lane  = tid & 31;
  const int wv    = tid >> 5;   // 0..7
  const int wm    = wv & 3;     // 4 waves along M (32 rows each)
  const int wn    = wv >> 2;    // 2 waves along N (32 cols each)

  v8f c[2][2];
#pragma unroll
  for (int mi = 0; mi < 2; ++mi)
#pragma unroll
    for (int ni = 0; ni < 2; ++ni)
#pragma unroll
      for (int r = 0; r < 8; ++r) c[mi][ni][r] = 0.0f;

  for (int kb = 0; kb < DD; kb += 32) {
    // ---- stage A tile (f32 -> bf16) ----
    {
      int r0 = tid >> 3;            // 0..31
      int kq = (tid & 7) * 4;       // 0..28 step 4
#pragma unroll
      for (int p = 0; p < 4; ++p) {
        int r = r0 + 32 * p;
        int m = mBase + r;
        float ax = 0.f, ay = 0.f, az = 0.f, aw = 0.f;
        if (m < NN) {
          int gk = kb + kq;
          const float* ap = A0 + (long)m * DD + gk;
          float x0 = ap[0], x1 = ap[1], x2 = ap[2], x3 = ap[3];
          if (MODE == 0) {
            const float* gp = A1 + (long)m * DD + gk;
            ax = x0 + gp[0]; ay = x1 + gp[1]; az = x2 + gp[2]; aw = x3 + gp[3];
          } else {
            ax = (x0 - mean[gk + 0]) * istd[gk + 0] * gamma[gk + 0] + beta[gk + 0];
            ay = (x1 - mean[gk + 1]) * istd[gk + 1] * gamma[gk + 1] + beta[gk + 1];
            az = (x2 - mean[gk + 2]) * istd[gk + 2] * gamma[gk + 2] + beta[gk + 2];
            aw = (x3 - mean[gk + 3]) * istd[gk + 3] * gamma[gk + 3] + beta[gk + 3];
          }
        }
        As[r][kq + 0] = (__bf16)ax;
        As[r][kq + 1] = (__bf16)ay;
        As[r][kq + 2] = (__bf16)az;
        As[r][kq + 3] = (__bf16)aw;
      }
    }
    // ---- stage B tile (f32 -> bf16, transposed to [n][k]) ----
    {
      int n  = tid & 63;
      int k0 = tid >> 6;  // 0..3
#pragma unroll
      for (int p = 0; p < 8; ++p) {
        int k = k0 + 4 * p;
        Bs[n][k] = (__bf16)W[(long)(kb + k) * DD + (nBase + n)];
      }
    }
    __syncthreads();

    // ---- build fragments per the CDNA5 16-bit VGPR layouts ----
    const int mrow0 = wm * 32 + (lane & 15);
    const int koffA = (lane >> 4) * 8;    // lanes 0-15: K 0..7/16..23 ; 16-31: 8..15/24..31
    const int ncol0 = wn * 32 + (lane & 15);
    const int koffB = (lane >> 4) * 8;    // lanes 0-15: K 0..7/16..23 ; 16-31: 8..15/24..31

    v16bf a[2], b[2];
#pragma unroll
    for (int mi = 0; mi < 2; ++mi) {
      float4* fp = reinterpret_cast<float4*>(&a[mi]);
      fp[0] = *reinterpret_cast<const float4*>(&As[mrow0 + 16 * mi][koffA]);
      fp[1] = *reinterpret_cast<const float4*>(&As[mrow0 + 16 * mi][koffA + 16]);
    }
#pragma unroll
    for (int ni = 0; ni < 2; ++ni) {
      float4* fp = reinterpret_cast<float4*>(&b[ni]);
      fp[0] = *reinterpret_cast<const float4*>(&Bs[ncol0 + 16 * ni][koffB]);
      fp[1] = *reinterpret_cast<const float4*>(&Bs[ncol0 + 16 * ni][koffB + 16]);
    }

#pragma unroll
    for (int mi = 0; mi < 2; ++mi)
#pragma unroll
      for (int ni = 0; ni < 2; ++ni)
        c[mi][ni] = __builtin_amdgcn_wmma_f32_16x16x32_bf16(
            false, a[mi], false, b[ni], (short)0, c[mi][ni], false, false);

    __syncthreads();
  }

  // ---- epilogue: bias, store, per-column BN statistics ----
  const int mrowLane = wm * 32 + ((lane < 16) ? 0 : 8);
#pragma unroll
  for (int ni = 0; ni < 2; ++ni) {
    int nGlob = nBase + wn * 32 + ni * 16 + (lane & 15);
    float bv = bias[nGlob];
    float s = 0.f, s2 = 0.f;
#pragma unroll
    for (int mi = 0; mi < 2; ++mi) {
#pragma unroll
      for (int r = 0; r < 8; ++r) {
        int mGlob = mBase + mrowLane + mi * 16 + r;
        if (mGlob < NN) {
          float v = c[mi][ni][r] + bv;
          Y[(long)mGlob * DD + nGlob] = v;
          s += v;
          s2 += v * v;
        }
      }
    }
    unsafeAtomicAdd(&colsum[nGlob], s);
    unsafeAtomicAdd(&colsumsq[nGlob], s2);
  }
}

// ---------------- BN stats finalize ----------------
__global__ void bn_finalize_kernel(const float* __restrict__ sum,
                                   const float* __restrict__ sumsq,
                                   float* __restrict__ mean,
                                   float* __restrict__ istd) {
  int col = threadIdx.x;
  if (col < DD) {
    float m = sum[col] * (1.0f / NN);
    float v = sumsq[col] * (1.0f / NN) - m * m;
    mean[col] = m;
    istd[col] = rsqrtf(v + BN_EPS);
  }
}

// ---------------- BN apply (in place) + layer readout ----------------
// <<<256,256>>> : block b handles rows b, b+256, ... for all 256 cols
__global__ __launch_bounds__(256)
void bn_apply_readout_kernel(float* __restrict__ y,
                             const float* __restrict__ mean,
                             const float* __restrict__ istd,
                             const float* __restrict__ gamma,
                             const float* __restrict__ beta,
                             float* __restrict__ repr) {
  int col = threadIdx.x;
  int rstart = blockIdx.x;
  float m = mean[col], is = istd[col], g = gamma[col], b = beta[col];
  float s = 0.f;
  for (int row = rstart; row < NN; row += 256) {
    long idx = (long)row * DD + col;
    float v = (y[idx] - m) * is * g + b;
    y[idx] = v;
    s += v;
  }
  unsafeAtomicAdd(&repr[col], s);
}

// ---------------- classifier: [1,1280] @ [1280,10] + bc ----------------
__global__ void classifier_kernel(const float* __restrict__ summary,
                                  const float* __restrict__ Wc,
                                  const float* __restrict__ bc,
                                  float* __restrict__ out) {
  int ccol = threadIdx.x;
  if (ccol < NC) {
    float acc = bc[ccol];
    for (int k = 0; k < NL * DD; ++k) acc += summary[k] * Wc[k * NC + ccol];
    out[ccol] = acc;
  }
}

extern "C" void kernel_launch(void* const* d_in, const int* in_sizes, int n_in,
                              void* d_out, int out_size, void* d_ws, size_t ws_size,
                              hipStream_t stream) {
  const float*     x   = (const float*)d_in[0];
  const long long* ei  = (const long long*)d_in[1];
  const float*     W1  = (const float*)d_in[2];
  const float*     b1  = (const float*)d_in[3];
  const float*     g1  = (const float*)d_in[4];
  const float*     be1 = (const float*)d_in[5];
  const float*     W2  = (const float*)d_in[6];
  const float*     b2  = (const float*)d_in[7];
  const float*     g2  = (const float*)d_in[8];
  const float*     be2 = (const float*)d_in[9];
  const float*     Wc  = (const float*)d_in[10];
  const float*     bc  = (const float*)d_in[11];

  const size_t BIG = (size_t)NN * DD;  // 12.8M floats
  float* hbuf = (float*)d_ws;          // also y2 (aliased; h dead when y2 written)
  float* agg  = hbuf + BIG;
  float* y1   = agg + BIG;
  float* st   = y1 + BIG;
  float* sum1 = st;         float* sumsq1 = st + 256;
  float* mean1 = st + 512;  float* istd1  = st + 768;
  float* sum2 = st + 1024;  float* sumsq2 = st + 1280;
  float* mean2 = st + 1536; float* istd2  = st + 1792;
  float* summary = st + 2048;  // 1280 floats

  if (ws_size < (3 * BIG + 2048 + NL * DD) * sizeof(float)) return;

  const long long* src = ei;
  const long long* dst = ei + NE;

  zero_f32<<<4, 256, 0, stream>>>(summary, NL * DD);

  dim3 gridG((NN + 127) / 128, DD / 64);
  const float* hcur = x;

  for (int i = 0; i < NL; ++i) {
    zero_f32<<<2048, 256, 0, stream>>>(agg, (long)BIG);
    zero_f32<<<2, 256, 0, stream>>>(sum1, 512);
    zero_f32<<<2, 256, 0, stream>>>(sum2, 512);

    scatter_add_kernel<<<(NE + 7) / 8, 256, 0, stream>>>(hcur, src, dst, agg);

    gin_gemm_kernel<0><<<gridG, 256, 0, stream>>>(
        hcur, agg, nullptr, nullptr, nullptr, nullptr,
        W1 + (size_t)i * DD * DD, b1 + i * DD, y1, sum1, sumsq1);

    bn_finalize_kernel<<<1, 256, 0, stream>>>(sum1, sumsq1, mean1, istd1);

    gin_gemm_kernel<1><<<gridG, 256, 0, stream>>>(
        y1, nullptr, mean1, istd1, g1 + i * DD, be1 + i * DD,
        W2 + (size_t)i * DD * DD, b2 + i * DD, hbuf, sum2, sumsq2);

    bn_finalize_kernel<<<1, 256, 0, stream>>>(sum2, sumsq2, mean2, istd2);

    bn_apply_readout_kernel<<<256, 256, 0, stream>>>(
        hbuf, mean2, istd2, g2 + i * DD, be2 + i * DD, summary + i * DD);

    hcur = hbuf;
  }

  classifier_kernel<<<1, 32, 0, stream>>>(summary, Wc, bc, (float*)d_out);
}
